// JIIF_Cou_36859409334297
// MI455X (gfx1250) — matrix operand order repaired and checked by reading, not compiled, not run on hardware
//
#include <hip/hip_runtime.h>
#include <hip/hip_bf16.h>

typedef __attribute__((ext_vector_type(16))) _Float16 v16h;
typedef __attribute__((ext_vector_type(8)))  _Float16 v8h;
typedef __attribute__((ext_vector_type(8)))  float    v8f;

#define B_    2
#define N_    16384
#define LH    64
#define LW    64
#define HH    256
#define HWID  256
#define CF    128
#define CG    128
#define BN    (B_ * N_)      // 32768
#define MROWS (4 * BN)       // 131072
#define MT    64             // rows per workgroup tile
#define K0P   416            // 386 padded to multiple of 32

// f16 weight buffer offsets (in halves) inside workspace
#define W0_OFF 0
#define W1_OFF (W0_OFF + 1024 * 416)
#define W2_OFF (W1_OFF + 512 * 1024)
#define W3_OFF (W2_OFF + 256 * 512)
#define W4_OFF (W3_OFF + 128 * 256)
#define WTOT   (W4_OFF + 16 * 128)

// ---------------------------------------------------------------------------
// Convert f32 weights [Nout, Kin] -> zero-padded f16 [NoutPad, Kpad]
// ---------------------------------------------------------------------------
__global__ void cvt_w_kernel(const float* __restrict__ src, _Float16* __restrict__ dst,
                             int Nout, int Kin, int NoutPad, int Kpad) {
    int total = NoutPad * Kpad;
    for (int i = blockIdx.x * blockDim.x + threadIdx.x; i < total;
         i += gridDim.x * blockDim.x) {
        int r = i / Kpad;
        int c = i - r * Kpad;
        float v = (r < Nout && c < Kin) ? src[r * Kin + c] : 0.0f;
        dst[i] = (_Float16)v;
    }
}

// ---------------------------------------------------------------------------
// Fragment loaders matching the CDNA5 16-bit WMMA VGPR layouts.
// A (16x32, MxK): lane<16 -> K 0..7 & 16..23; lane>=16 -> K 8..15 & 24..31.
// B (32x16, KxN): lane<16 -> K 0..7 & 16..23; lane>=16 -> K 8..15 & 24..31 (same as A).
// ---------------------------------------------------------------------------
__device__ __forceinline__ v16h load_a_frag(const _Float16* p) {
    v8h lo = *(const v8h*)(p);        // ds_load_b128
    v8h hi = *(const v8h*)(p + 16);   // ds_load_b128
    v16h a;
#pragma unroll
    for (int i = 0; i < 8; ++i) { a[i] = lo[i]; a[i + 8] = hi[i]; }
    return a;
}
__device__ __forceinline__ v16h load_b_frag(const _Float16* p) {
    v8h lo = *(const v8h*)(p);        // global_load_b128
    v8h hi = *(const v8h*)(p + 16);   // global_load_b128
    v16h b;
#pragma unroll
    for (int i = 0; i < 8; ++i) { b[i] = lo[i]; b[i + 8] = hi[i]; }
    return b;
}
#define WMMA_F16(A, Bf, C) \
    __builtin_amdgcn_wmma_f32_16x16x32_f16(false, (A), false, (Bf), (short)0, (C), false, false)

// ---------------------------------------------------------------------------
// Hidden MLP layer on a 64-row LDS tile, 2x2 register-blocked WMMA.
// Each wave owns 32x32 output blocks: 4 independent accumulators, and per
// K-step only 2 A-frags + 2 B-frags feed 4 WMMAs (half the load traffic,
// 4 independent WMMA dependency chains).
// ---------------------------------------------------------------------------
__device__ __forceinline__ void mlp_layer_wmma(
    const _Float16* __restrict__ Wl, const float* __restrict__ bias,
    const _Float16* bufIn, _Float16* bufOut,
    int Kpad, int Nout, bool relu) {

    const int lane = threadIdx.x & 31;
    const int wave = threadIdx.x >> 5;
    const int rowL = lane & 15;   // M (A) / N (B,C,D) within 16-tile
    const int hk   = lane >> 4;   // half-K selector

    const int nNB = Nout >> 5;            // 32-wide N blocks
    const int totalBlocks = 2 * nNB;      // (64/32) x (Nout/32)

    for (int t = wave; t < totalBlocks; t += 8) {
        const int mb = t & 1;
        const int nb = t >> 1;

        v8f acc00 = {}, acc01 = {}, acc10 = {}, acc11 = {};
        const int ksteps = Kpad >> 5;
        const _Float16* aBase0 = bufIn + (mb * 32 + rowL) * Kpad + hk * 8;
        const _Float16* aBase1 = aBase0 + 16 * Kpad;
        const _Float16* bBase0 = Wl + (size_t)(nb * 32 + rowL) * Kpad + hk * 8;
        const _Float16* bBase1 = bBase0 + (size_t)16 * Kpad;

        for (int kk = 0; kk < ksteps; ++kk) {
            const v16h a0 = load_a_frag(aBase0 + kk * 32);
            const v16h a1 = load_a_frag(aBase1 + kk * 32);
            const v16h bf0 = load_b_frag(bBase0 + kk * 32);
            const v16h bf1 = load_b_frag(bBase1 + kk * 32);
            acc00 = WMMA_F16(a0, bf0, acc00);
            acc01 = WMMA_F16(a0, bf1, acc01);
            acc10 = WMMA_F16(a1, bf0, acc10);
            acc11 = WMMA_F16(a1, bf1, acc11);
        }

        // Epilogue: D layout -> VGPR i holds M = i (+8 for lanes 16..31), N = lane%16
#pragma unroll
        for (int mi = 0; mi < 2; ++mi) {
#pragma unroll
            for (int ni = 0; ni < 2; ++ni) {
                const v8f acc = (mi == 0) ? (ni == 0 ? acc00 : acc01)
                                          : (ni == 0 ? acc10 : acc11);
                const int col = nb * 32 + ni * 16 + rowL;
                const float bv = bias[col];
#pragma unroll
                for (int i = 0; i < 8; ++i) {
                    const int rowOut = mb * 32 + mi * 16 + hk * 8 + i;
                    float v = acc[i] + bv;
                    if (relu) v = fmaxf(v, 0.0f);
                    bufOut[rowOut * Nout + col] = (_Float16)v;
                }
            }
        }
    }
}

// ---------------------------------------------------------------------------
// Final layer: [64,128] x [16,128]^T -> per-row (pred, logit) f32 to global.
// Only 4 output tiles; waves 0..3 each take one.
// ---------------------------------------------------------------------------
__device__ __forceinline__ void mlp_final_wmma(
    const _Float16* __restrict__ Wl, const float* __restrict__ bias,
    const _Float16* bufIn, float* __restrict__ preds, int mBase) {

    const int lane = threadIdx.x & 31;
    const int wave = threadIdx.x >> 5;
    const int rowL = lane & 15;
    const int hk   = lane >> 4;

    if (wave < 4) {
        const int mi = wave;
        v8f acc = {};
        const _Float16* aBase = bufIn + (mi * 16 + rowL) * 128 + hk * 8;
        const _Float16* bBase = Wl + (size_t)rowL * 128 + hk * 8;
#pragma unroll
        for (int kk = 0; kk < 4; ++kk) {
            const v16h a = load_a_frag(aBase + kk * 32);
            const v16h bf = load_b_frag(bBase + kk * 32);
            acc = WMMA_F16(a, bf, acc);
        }
        const int col = rowL;
        const float bv = (col < 2) ? bias[col] : 0.0f;
#pragma unroll
        for (int i = 0; i < 8; ++i) {
            const int rowOut = mi * 16 + hk * 8 + i;
            if (col < 2) preds[(size_t)(mBase + rowOut) * 2 + col] = acc[i] + bv;
        }
    }
}

// ---------------------------------------------------------------------------
// Fused: gather 64 rows of the 386-dim MLP input into LDS, run 5 layers.
// ---------------------------------------------------------------------------
__global__ void jiif_fused_kernel(
    const float* __restrict__ feat, const float* __restrict__ coord,
    const float* __restrict__ hr_guide, const float* __restrict__ lr_guide,
    const _Float16* __restrict__ wbuf,
    const float* __restrict__ b0, const float* __restrict__ b1,
    const float* __restrict__ b2, const float* __restrict__ b3,
    const float* __restrict__ b4,
    float* __restrict__ preds) {

    extern __shared__ _Float16 smem[];
    _Float16* buf0 = smem;            // capacity MT x 512  (in: 416, outs: 512/128)
    _Float16* buf1 = smem + MT * 512; // capacity MT x 1024 (outs: 1024/256)

    const int mBase = blockIdx.x * MT;
    const int lane = threadIdx.x & 31;
    const int wave = threadIdx.x >> 5;

    // ---- gather phase: each wave builds 8 rows ----
    for (int rr = 0; rr < 8; ++rr) {
        const int r = wave * 8 + rr;
        const int m = mBase + r;
        const int k   = m >> 15;        // / BN
        const int rem = m & 32767;
        const int b   = rem >> 14;      // / N_
        const int n   = rem & 16383;

        const float cy = coord[(size_t)(b * N_ + n) * 2 + 0];
        const float cx = coord[(size_t)(b * N_ + n) * 2 + 1];

        // nearest sample in HR guide
        const float riy = rintf((cy + 1.0f) * (0.5f * HH) - 0.5f);
        const float rix = rintf((cx + 1.0f) * (0.5f * HWID) - 0.5f);
        const bool validH = (riy >= 0.0f) && (riy < (float)HH) &&
                            (rix >= 0.0f) && (rix < (float)HWID);
        const int iyH = min(max((int)riy, 0), HH - 1);
        const int ixH = min(max((int)rix, 0), HWID - 1);

        // offset sample in LR grids (offs[k] scaled by [1/h, 1/w])
        const float offy = ((k & 2) ? 1.0f : -1.0f) * (1.0f / LH);
        const float offx = ((k & 1) ? 1.0f : -1.0f) * (1.0f / LW);
        const float cy2 = cy + offy, cx2 = cx + offx;
        const float riy2 = rintf((cy2 + 1.0f) * (0.5f * LH) - 0.5f);
        const float rix2 = rintf((cx2 + 1.0f) * (0.5f * LW) - 0.5f);
        const bool validL = (riy2 >= 0.0f) && (riy2 < (float)LH) &&
                            (rix2 >= 0.0f) && (rix2 < (float)LW);
        const int iyL = min(max((int)riy2, 0), LH - 1);
        const int ixL = min(max((int)rix2, 0), LW - 1);

        const float qcy = validL ? (-1.0f + 2.0f * ((float)iyL + 0.5f) / (float)LH) : 0.0f;
        const float qcx = validL ? (-1.0f + 2.0f * ((float)ixL + 0.5f) / (float)LW) : 0.0f;
        const float rely = (cy - qcy) * (float)LH;
        const float relx = (cx - qcx) * (float)LW;

        const float* hrB = hr_guide + (size_t)b * CG * HH * HWID + iyH * HWID + ixH;
        const float* ftB = feat     + (size_t)b * CF * LH * LW   + iyL * LW   + ixL;
        const float* lrB = lr_guide + (size_t)b * CG * LH * LW   + iyL * LW   + ixL;

        for (int c = lane; c < K0P; c += 32) {
            float v;
            if (c < 128) {
                v = validL ? ftB[(size_t)c * LH * LW] : 0.0f;
            } else if (c < 256) {
                v = validH ? hrB[(size_t)(c - 128) * HH * HWID] : 0.0f;
            } else if (c < 384) {
                const int cc = c - 256;
                const float g = validH ? hrB[(size_t)cc * HH * HWID] : 0.0f;
                const float l = validL ? lrB[(size_t)cc * LH * LW] : 0.0f;
                v = g - l;
            } else if (c == 384) { v = rely; }
            else if (c == 385)   { v = relx; }
            else                 { v = 0.0f; }
            buf0[r * K0P + c] = (_Float16)v;
        }
    }
    __syncthreads();

    // ---- MLP: 386(pad416) -> 1024 -> 512 -> 256 -> 128 -> 2(pad16) ----
    mlp_layer_wmma(wbuf + W0_OFF, b0, buf0, buf1, 416, 1024, true);
    __syncthreads();
    mlp_layer_wmma(wbuf + W1_OFF, b1, buf1, buf0, 1024, 512, true);
    __syncthreads();
    mlp_layer_wmma(wbuf + W2_OFF, b2, buf0, buf1, 512, 256, true);
    __syncthreads();
    mlp_layer_wmma(wbuf + W3_OFF, b3, buf1, buf0, 256, 128, true);
    __syncthreads();
    mlp_final_wmma(wbuf + W4_OFF, b4, buf0, preds, mBase);
}

// ---------------------------------------------------------------------------
// Softmax over the 4 offset candidates, weighted sum -> [B,N,1]
// ---------------------------------------------------------------------------
__global__ void softmax_combine_kernel(const float* __restrict__ preds,
                                       float* __restrict__ out) {
    const int idx = blockIdx.x * blockDim.x + threadIdx.x;
    if (idx >= BN) return;
    float p0[4], p1[4];
#pragma unroll
    for (int k = 0; k < 4; ++k) {
        const size_t m = (size_t)k * BN + idx;
        p0[k] = preds[m * 2 + 0];
        p1[k] = preds[m * 2 + 1];
    }
    const float mx = fmaxf(fmaxf(p1[0], p1[1]), fmaxf(p1[2], p1[3]));
    float s = 0.0f, acc = 0.0f;
#pragma unroll
    for (int k = 0; k < 4; ++k) {
        const float e = __expf(p1[k] - mx);
        s += e;
        acc += p0[k] * e;
    }
    out[idx] = acc / s;
}

// ---------------------------------------------------------------------------
extern "C" void kernel_launch(void* const* d_in, const int* in_sizes, int n_in,
                              void* d_out, int out_size, void* d_ws, size_t ws_size,
                              hipStream_t stream) {
    (void)in_sizes; (void)n_in; (void)out_size; (void)ws_size;

    const float* feat     = (const float*)d_in[0];
    const float* coord    = (const float*)d_in[1];
    const float* hr_guide = (const float*)d_in[2];
    const float* lr_guide = (const float*)d_in[3];
    const float* W0 = (const float*)d_in[4];  const float* b0 = (const float*)d_in[5];
    const float* W1 = (const float*)d_in[6];  const float* b1 = (const float*)d_in[7];
    const float* W2 = (const float*)d_in[8];  const float* b2 = (const float*)d_in[9];
    const float* W3 = (const float*)d_in[10]; const float* b3 = (const float*)d_in[11];
    const float* W4 = (const float*)d_in[12]; const float* b4 = (const float*)d_in[13];

    _Float16* wbuf = (_Float16*)d_ws;
    float* preds = (float*)((char*)d_ws + (size_t)WTOT * sizeof(_Float16));

    // f32 -> padded f16 weights (recomputed each call; deterministic)
    cvt_w_kernel<<<512, 256, 0, stream>>>(W0, wbuf + W0_OFF, 1024, 386, 1024, 416);
    cvt_w_kernel<<<512, 256, 0, stream>>>(W1, wbuf + W1_OFF, 512, 1024, 512, 1024);
    cvt_w_kernel<<<256, 256, 0, stream>>>(W2, wbuf + W2_OFF, 256, 512, 256, 512);
    cvt_w_kernel<<<64,  256, 0, stream>>>(W3, wbuf + W3_OFF, 128, 256, 128, 256);
    cvt_w_kernel<<<8,   256, 0, stream>>>(W4, wbuf + W4_OFF, 2, 128, 16, 128);

    const size_t ldsBytes = (size_t)(MT * 512 + MT * 1024) * sizeof(_Float16); // 192 KB
    static_assert((MT * 512 + MT * 1024) * sizeof(_Float16) <= 320 * 1024, "LDS fits WGP");
    hipFuncSetAttribute((const void*)jiif_fused_kernel,
                        hipFuncAttributeMaxDynamicSharedMemorySize, (int)ldsBytes);

    jiif_fused_kernel<<<MROWS / MT, 256, ldsBytes, stream>>>(
        feat, coord, hr_guide, lr_guide, wbuf, b0, b1, b2, b3, b4, preds);

    softmax_combine_kernel<<<(BN + 255) / 256, 256, 0, stream>>>(preds, (float*)d_out);
}
